// FractalSSMLayer_31877247270953
// MI455X (gfx1250) — matrix operand order repaired and checked by reading, not compiled, not run on hardware
//
#include <hip/hip_runtime.h>

// ---------------------------------------------------------------------------
// FractalSSMLayer for MI455X (gfx1250, wave32, WMMA bf16 16x16x32)
// BATCH=8, SEQ=2048, D=1024, N=16, DT_RANK=64
// ---------------------------------------------------------------------------
#define BATCHN 8
#define SEQL   2048
#define DDIM   1024
#define NST    16
#define DTRANK 64
#define MTOT   (BATCHN * SEQL)   /* 16384 rows */
#define KAPPA_C 0.5f

typedef __attribute__((ext_vector_type(16))) __bf16 v16bf;
typedef __attribute__((ext_vector_type(8)))  float  v8f;

// CDNA5 async global->LDS path (guarded: falls back to vector ld/st if absent)
#if defined(__AMDGCN__) && __has_builtin(__builtin_amdgcn_global_load_async_to_lds_b128) && __has_builtin(__builtin_amdgcn_s_wait_asynccnt)
#define USE_ASYNC_LDS 1
// Builtin prototype (from clang diagnostic): arg0 = AS(1) int4* (non-const).
typedef int v4i_gcc __attribute__((vector_size(16)));
typedef __attribute__((address_space(1))) v4i_gcc* as1_v4p;
typedef __attribute__((address_space(3))) v4i_gcc* as3_v4p;
#else
#define USE_ASYNC_LDS 0
#endif

__device__ __forceinline__ unsigned short f2bf(float f) {
  unsigned int u = __builtin_bit_cast(unsigned int, f);
  u += 0x7FFFu + ((u >> 16) & 1u);          // round-to-nearest-even
  return (unsigned short)(u >> 16);
}
__device__ __forceinline__ float silu_f(float v) {
  return v / (1.0f + __expf(-v));
}

union V16U {            // 32B: two b128 LDS loads -> one WMMA bf16 operand
  uint4 q[2];
  v16bf v;
};

// ----------------- f32 -> bf16, transposed (dst is N x K) ------------------
__global__ __launch_bounds__(256) void fssm_cvt_bf16_t(
    const float* __restrict__ src, unsigned short* __restrict__ dst,
    int K, int N) {
  int i = blockIdx.x * 256 + threadIdx.x;
  if (i < K * N) {
    int k = i / N, n = i % N;
    dst[(size_t)n * K + k] = f2bf(src[i]);
  }
}

// ------------------------------ LayerNorm ----------------------------------
__global__ __launch_bounds__(256) void fssm_layernorm_bf16(
    const float* __restrict__ x, const float* __restrict__ w,
    const float* __restrict__ b, unsigned short* __restrict__ out) {
  __shared__ float s1[256], s2[256];
  const int row = blockIdx.x;
  const float* xr = x + (size_t)row * DDIM;
  float a = 0.f, q = 0.f;
#pragma unroll
  for (int i = 0; i < 4; ++i) {
    float v = xr[threadIdx.x + i * 256];
    a += v; q += v * v;
  }
  s1[threadIdx.x] = a; s2[threadIdx.x] = q;
  __syncthreads();
  for (int s = 128; s > 0; s >>= 1) {
    if (threadIdx.x < s) {
      s1[threadIdx.x] += s1[threadIdx.x + s];
      s2[threadIdx.x] += s2[threadIdx.x + s];
    }
    __syncthreads();
  }
  float mu  = s1[0] * (1.0f / DDIM);
  float var = s2[0] * (1.0f / DDIM) - mu * mu;
  float inv = rsqrtf(var + 1e-5f);
#pragma unroll
  for (int i = 0; i < 4; ++i) {
    int c = threadIdx.x + i * 256;
    float v = (xr[c] - mu) * inv * w[c] + b[c];
    out[(size_t)row * DDIM + c] = f2bf(v);
  }
}

// ------------------------- WMMA bf16 GEMM (64x64 tile) ---------------------
// C[MxN] = A[MxK](bf16, row-major) * Bt[NxK](bf16, pre-transposed row-major)
// (+resid)(+bias), f32 out. 128 threads = 4 waves; wave w -> 16-row stripe,
// 4 col tiles of 16. M%64==0 and K%32==0 guaranteed; N guarded (clamped).
__global__ __launch_bounds__(128) void fssm_wmma_gemm_bf16(
    const unsigned short* __restrict__ A, const unsigned short* __restrict__ Bt,
    float* __restrict__ C, int M, int N, int K,
    const float* __restrict__ resid, const float* __restrict__ bias) {
  (void)M;
  __shared__ __align__(16) unsigned short As[64][48];  // 96B row stride
  __shared__ __align__(16) unsigned short Bs[64][48];  // Bs[col][k]

  const int tid   = threadIdx.x;
  const int lane  = tid & 31;
  const int wave  = tid >> 5;
  const int lrow  = lane & 15;
  const int khalf = lane >> 4;
  const int rowBase = blockIdx.y * 64;
  const int colBase = blockIdx.x * 64;

  // cooperative staging: 64 rows x 2 segments of 16 bf16 (32B) each
  const int srow = tid >> 1, sseg = tid & 1;

  int bRowG = colBase + srow;
  if (bRowG >= N) bRowG = colBase;   // clamp for narrow N (N=32 case)

  v8f acc[4];
#pragma unroll
  for (int c = 0; c < 4; ++c)
#pragma unroll
    for (int e = 0; e < 8; ++e) acc[c][e] = 0.0f;

  for (int k0 = 0; k0 < K; k0 += 32) {
    __syncthreads();
    const unsigned short* asrc = A  + (size_t)(rowBase + srow) * K + k0 + sseg * 16;
    const unsigned short* bsrc = Bt + (size_t)bRowG * K            + k0 + sseg * 16;
    unsigned short* adst = &As[srow][sseg * 16];
    unsigned short* bdst = &Bs[srow][sseg * 16];
#if USE_ASYNC_LDS
    // gfx1250 async DMA: 16B/lane global->LDS, tracked by ASYNCcnt
    __builtin_amdgcn_global_load_async_to_lds_b128(
        (as1_v4p)asrc, (as3_v4p)adst, 0, 0);
    __builtin_amdgcn_global_load_async_to_lds_b128(
        (as1_v4p)(asrc + 8), (as3_v4p)(adst + 8), 0, 0);
    __builtin_amdgcn_global_load_async_to_lds_b128(
        (as1_v4p)bsrc, (as3_v4p)bdst, 0, 0);
    __builtin_amdgcn_global_load_async_to_lds_b128(
        (as1_v4p)(bsrc + 8), (as3_v4p)(bdst + 8), 0, 0);
#else
    {
      uint4 a0 = ((const uint4*)asrc)[0];
      uint4 a1 = ((const uint4*)asrc)[1];
      uint4 b0 = ((const uint4*)bsrc)[0];
      uint4 b1 = ((const uint4*)bsrc)[1];
      *(uint4*)adst       = a0;
      *(uint4*)(adst + 8) = a1;
      *(uint4*)bdst       = b0;
      *(uint4*)(bdst + 8) = b1;
    }
#endif
    if (k0 + 32 < K) {  // global_prefetch_b8 of next K tile
      __builtin_prefetch(asrc + 32, 0, 1);
      __builtin_prefetch(bsrc + 32, 0, 1);
    }
#if USE_ASYNC_LDS
    __builtin_amdgcn_s_wait_asynccnt(0);
#endif
    __syncthreads();

    // A fragment: elem e -> K = khalf*16 + e (contiguous)
    V16U ua;
    ua.q[0] = *(const uint4*)&As[wave * 16 + lrow][khalf * 16];
    ua.q[1] = *(const uint4*)&As[wave * 16 + lrow][khalf * 16 + 8];
#pragma unroll
    for (int c = 0; c < 4; ++c) {
      // B fragment: lane -> column, elem e -> K = khalf*16 + e (contiguous)
      V16U ub;
      ub.q[0] = *(const uint4*)&Bs[c * 16 + lrow][khalf * 16];
      ub.q[1] = *(const uint4*)&Bs[c * 16 + lrow][khalf * 16 + 8];
      acc[c] = __builtin_amdgcn_wmma_f32_16x16x32_bf16(
          false, ua.v, false, ub.v, (short)0, acc[c], false, false);
    }
  }

  // store: VGPR r -> M = r + 8*khalf within the wave's 16-row stripe
  const int crow = rowBase + wave * 16 + khalf * 8;
#pragma unroll
  for (int c = 0; c < 4; ++c) {
    int col = colBase + c * 16 + lrow;
    if (col < N) {
#pragma unroll
      for (int r = 0; r < 8; ++r) {
        size_t idx = (size_t)(crow + r) * N + col;
        float v = acc[c][r];
        if (resid) v += resid[idx];
        if (bias)  v += bias[col];
        C[idx] = v;
      }
    }
  }
}

// -------------------------- elementwise kernels ----------------------------
__global__ __launch_bounds__(256) void fssm_silu_split(
    const float* __restrict__ xz, float* __restrict__ xin_f,
    unsigned short* __restrict__ xin_bf) {
  size_t i = (size_t)blockIdx.x * 256 + threadIdx.x;
  if (i >= (size_t)MTOT * DDIM) return;
  size_t row = i >> 10, col = i & 1023;
  float v = silu_f(xz[row * (2 * DDIM) + col]);
  xin_f[i]  = v;
  xin_bf[i] = f2bf(v);
}

__global__ __launch_bounds__(256) void fssm_slice_dtr_bf16(
    const float* __restrict__ xp, unsigned short* __restrict__ dtr_bf) {
  size_t i = (size_t)blockIdx.x * 256 + threadIdx.x;
  if (i >= (size_t)MTOT * DTRANK) return;
  size_t row = i >> 6, col = i & 63;
  dtr_bf[i] = f2bf(xp[row * 128 + col]);
}

__global__ __launch_bounds__(256) void fssm_softplus_bias(
    float* __restrict__ dt, const float* __restrict__ b_dt) {
  size_t i = (size_t)blockIdx.x * 256 + threadIdx.x;
  if (i >= (size_t)MTOT * DDIM) return;
  float v = dt[i] + b_dt[i & 1023];
  dt[i] = (v > 20.0f) ? v : log1pf(__expf(v));
}

__global__ __launch_bounds__(256) void fssm_gate(
    const float* __restrict__ y, const float* __restrict__ xz,
    unsigned short* __restrict__ yg_bf) {
  size_t i = (size_t)blockIdx.x * 256 + threadIdx.x;
  if (i >= (size_t)MTOT * DDIM) return;
  size_t row = i >> 10, col = i & 1023;
  float g = silu_f(xz[row * (2 * DDIM) + DDIM + col]);
  yg_bf[i] = f2bf(y[i] * g);
}

// ------------------------------- the scan ----------------------------------
// 1 block per batch, 1024 threads (32 wave32s). Thread d owns channel d:
// h[d][n] in registers (16 complex). Per-step cross-D mean of h via
// shuffle + two-level LDS reduction (deterministic order).
// Transcendental-minimized: polar rescale trick + half-angle identity.
__global__ __launch_bounds__(1024) void fssm_scan(
    const float* __restrict__ x_in, const float* __restrict__ dt,
    const float* __restrict__ xp,   const float* __restrict__ x_res,
    const float* __restrict__ A_log, float* __restrict__ y,
    float* __restrict__ epart) {
  const int b = blockIdx.x;
  const int d = threadIdx.x;
  const int lane = d & 31, wave = d >> 5;

  __shared__ float stage[96];          // B[32] | C[32] | r[32]
  __shared__ float wavebuf[32][32];    // [wave][j], j = n*2+c
  __shared__ float hmsum[32];
  __shared__ float Tsh[NST];
  __shared__ float ered[32];

  float Aneg[NST];
#pragma unroll
  for (int n = 0; n < NST; ++n) Aneg[n] = -__expf(A_log[n]);

  float h0[NST], h1[NST];
#pragma unroll
  for (int n = 0; n < NST; ++n) {      // phases = 2*pi*n/16
    float ph = 0.392699081698724f * (float)n;   // pi/8 * n
    h0[n] = 0.01f * __cosf(ph);
    h1[n] = 0.01f * __sinf(ph);
  }
  float eacc = 0.f;

  for (int t = 0; t < SEQL; ++t) {
    __syncthreads();                   // protect stage/Tsh from prior step
    const size_t rowbd = (size_t)b * SEQL + t;
    if (d < 64)      stage[d] = xp[rowbd * 128 + 64 + d];        // B then C
    else if (d < 96) stage[d] = x_res[rowbd * 32 + (d - 64)];    // r_t

    float xd  = x_in[rowbd * DDIM + d];
    float dtd = dt[rowbd * DDIM + d];
    float xdt = xd * dtd;

    // ---- reduce h over D (pre-update state), j = 2n (+1) ----
#pragma unroll
    for (int n = 0; n < NST; ++n) {
      float v0 = h0[n], v1 = h1[n];
#pragma unroll
      for (int s = 16; s > 0; s >>= 1) {
        v0 += __shfl_xor(v0, s, 32);
        v1 += __shfl_xor(v1, s, 32);
      }
      if (lane == 0) { wavebuf[wave][2 * n] = v0; wavebuf[wave][2 * n + 1] = v1; }
    }
    __syncthreads();
    if (wave == 0) {                   // lane j sums column j across waves
      float s = 0.f;
#pragma unroll
      for (int w = 0; w < 32; ++w) s += wavebuf[w][lane];
      hmsum[lane] = s;                 // mean scale cancels in T
    }
    __syncthreads();
    if (d < NST) {
      // T = cos^2((phi_r - phi_m)/2) = 0.5*(1 + r.m / (|r||m|))
      float m0 = hmsum[2 * d], m1 = hmsum[2 * d + 1];
      float r0 = stage[64 + 2 * d], r1 = stage[64 + 2 * d + 1];
      float dot = r0 * m0 + r1 * m1;
      float den = sqrtf((r0 * r0 + r1 * r1) * (m0 * m0 + m1 * m1)) + 1e-12f;
      Tsh[d] = 0.5f * (1.f + dot / den);
    }
    __syncthreads();

    float yd = 0.f;
#pragma unroll
    for (int n = 0; n < NST; ++n) {
      float alpha = __expf(dtd * Aneg[n]);
      float expo  = 1.f + (1.f - alpha) * KAPPA_C * (1.f - Tsh[n]);
      float mag   = sqrtf(h0[n] * h0[n] + h1[n] * h1[n] + 1e-8f);
      float cmag  = fminf(__expf(expo * __logf(mag + 1e-8f)), 10.f);
      eacc += mag - cmag;
      float sc = cmag / mag;           // polar(cmag, phase(h)) == h * cmag/|h|
      float a0 = h0[n] * sc + stage[2 * n]     * xdt;
      float a1 = h1[n] * sc + stage[2 * n + 1] * xdt;
      float m2 = sqrtf(a0 * a0 + a1 * a1 + 1e-8f);
      float e2 = __expf(-2.f * m2);    // tanh via exp
      float s2 = ((1.f - e2) / (1.f + e2)) / m2;
      h0[n] = a0 * s2; h1[n] = a1 * s2;
      yd += stage[32 + 2 * n] * h0[n] + stage[32 + 2 * n + 1] * h1[n];
    }
    y[rowbd * DDIM + d] = yd;
  }

  // deterministic energy reduction
#pragma unroll
  for (int s = 16; s > 0; s >>= 1) eacc += __shfl_xor(eacc, s, 32);
  if (lane == 0) ered[wave] = eacc;
  __syncthreads();
  if (d == 0) {
    float s = 0.f;
#pragma unroll
    for (int w = 0; w < 32; ++w) s += ered[w];
    epart[b] = s;
  }
}

__global__ void fssm_energy_final(const float* __restrict__ ep, float* __restrict__ out) {
  if (threadIdx.x == 0 && blockIdx.x == 0) {
    float s = 0.f;
#pragma unroll
    for (int i = 0; i < BATCHN; ++i) s += ep[i];
    out[0] = s;
  }
}

// ------------------------------- launcher ----------------------------------
extern "C" void kernel_launch(void* const* d_in, const int* in_sizes, int n_in,
                              void* d_out, int out_size, void* d_ws, size_t ws_size,
                              hipStream_t stream) {
  (void)in_sizes; (void)n_in; (void)out_size; (void)ws_size;
  const float* x     = (const float*)d_in[0];
  const float* ln_w  = (const float*)d_in[1];
  const float* ln_b  = (const float*)d_in[2];
  const float* W_in  = (const float*)d_in[3];
  const float* W_xp  = (const float*)d_in[4];
  const float* W_dt  = (const float*)d_in[5];
  const float* b_dt  = (const float*)d_in[6];
  const float* W_res = (const float*)d_in[7];
  const float* W_out = (const float*)d_in[8];
  const float* b_out = (const float*)d_in[9];
  const float* A_log = (const float*)d_in[10];

  char* ws = (char*)d_ws;
  size_t cur = 0;
  auto carve = [&](size_t bytes) -> char* {
    char* p = ws + cur;
    cur = (cur + bytes + 255) & ~(size_t)255;
    return p;
  };
  // transposed bf16 weights (N x K layouts)
  unsigned short* wInT  = (unsigned short*)carve((size_t)DDIM * 2048 * 2);
  unsigned short* wXpT  = (unsigned short*)carve((size_t)DDIM * 128 * 2);
  unsigned short* wDtT  = (unsigned short*)carve((size_t)DTRANK * DDIM * 2);
  unsigned short* wResT = (unsigned short*)carve((size_t)DDIM * 32 * 2);
  unsigned short* wOutT = (unsigned short*)carve((size_t)DDIM * DDIM * 2);
  unsigned short* xnB   = (unsigned short*)carve((size_t)MTOT * DDIM * 2);
  float*          xz    = (float*)carve((size_t)MTOT * 2048 * 4);
  float*          xinF  = (float*)carve((size_t)MTOT * DDIM * 4);
  unsigned short* xinB  = (unsigned short*)carve((size_t)MTOT * DDIM * 2);
  float*          xpB   = (float*)carve((size_t)MTOT * 128 * 4);
  float*          xres  = (float*)carve((size_t)MTOT * 32 * 4);
  unsigned short* dtrB  = (unsigned short*)carve((size_t)MTOT * DTRANK * 2);
  float*          dtB   = (float*)carve((size_t)MTOT * DDIM * 4);
  float*          yB    = (float*)carve((size_t)MTOT * DDIM * 4);
  unsigned short* ygB   = (unsigned short*)carve((size_t)MTOT * DDIM * 2);
  float*          epart = (float*)carve(BATCHN * 4);

  auto nb = [](size_t n) { return (unsigned)((n + 255) / 256); };

  // 1) weights -> bf16, transposed to (N x K)
  fssm_cvt_bf16_t<<<nb((size_t)DDIM * 2048), 256, 0, stream>>>(W_in,  wInT,  DDIM, 2048);
  fssm_cvt_bf16_t<<<nb((size_t)DDIM * 128),  256, 0, stream>>>(W_xp,  wXpT,  DDIM, 128);
  fssm_cvt_bf16_t<<<nb((size_t)DTRANK * DDIM), 256, 0, stream>>>(W_dt, wDtT,  DTRANK, DDIM);
  fssm_cvt_bf16_t<<<nb((size_t)DDIM * 32),   256, 0, stream>>>(W_res, wResT, DDIM, 32);
  fssm_cvt_bf16_t<<<nb((size_t)DDIM * DDIM), 256, 0, stream>>>(W_out, wOutT, DDIM, DDIM);

  // 2) layernorm -> bf16
  fssm_layernorm_bf16<<<MTOT, 256, 0, stream>>>(x, ln_w, ln_b, xnB);

  // 3) xz = xn @ W_in   (16384 x 1024 x 2048)
  fssm_wmma_gemm_bf16<<<dim3(2048 / 64, MTOT / 64), 128, 0, stream>>>(
      xnB, wInT, xz, MTOT, 2048, DDIM, nullptr, nullptr);

  // 4) x_in = silu(xz[:, :1024])
  fssm_silu_split<<<nb((size_t)MTOT * DDIM), 256, 0, stream>>>(xz, xinF, xinB);

  // 5) xp = x_in @ W_xp  (N=128);  x_res = x_in @ W_res (N=32, guarded)
  fssm_wmma_gemm_bf16<<<dim3(128 / 64, MTOT / 64), 128, 0, stream>>>(
      xinB, wXpT, xpB, MTOT, 128, DDIM, nullptr, nullptr);
  fssm_wmma_gemm_bf16<<<dim3(1, MTOT / 64), 128, 0, stream>>>(
      xinB, wResT, xres, MTOT, 32, DDIM, nullptr, nullptr);

  // 6) dt = softplus(xp[:, :64] @ W_dt + b_dt)
  fssm_slice_dtr_bf16<<<nb((size_t)MTOT * DTRANK), 256, 0, stream>>>(xpB, dtrB);
  fssm_wmma_gemm_bf16<<<dim3(DDIM / 64, MTOT / 64), 128, 0, stream>>>(
      dtrB, wDtT, dtB, MTOT, DDIM, DTRANK, nullptr, nullptr);
  fssm_softplus_bias<<<nb((size_t)MTOT * DDIM), 256, 0, stream>>>(dtB, b_dt);

  // 7) sequential scan (1 workgroup of 32 waves per batch)
  fssm_scan<<<BATCHN, 1024, 0, stream>>>(xinF, dtB, xpB, xres, A_log, yB, epart);

  // 8) out = residual + (y * silu(z)) @ W_out + b_out
  fssm_gate<<<nb((size_t)MTOT * DDIM), 256, 0, stream>>>(yB, xz, ygB);
  fssm_wmma_gemm_bf16<<<dim3(DDIM / 64, MTOT / 64), 128, 0, stream>>>(
      ygB, wOutT, (float*)d_out, MTOT, DDIM, DDIM, x, b_out);

  // 9) energy scalar
  fssm_energy_final<<<1, 32, 0, stream>>>(epart, (float*)d_out + (size_t)MTOT * DDIM);
}
